// Attention_19670950216611
// MI455X (gfx1250) — compile-verified, matrix operand order checked by reading, not run on hardware
//
#include <hip/hip_runtime.h>
#include <hip/hip_bf16.h>

typedef __attribute__((ext_vector_type(16))) _Float16 v16h;
typedef __attribute__((ext_vector_type(8)))  float    v8f;
typedef __attribute__((ext_vector_type(4)))  float    f32x4;

#define B_ 8
#define S_ 512
#define F_ 128
#define OUTW (F_ + F_ * F_)   // 16512 floats per output row

union FragU { v16h h; f32x4 f4[2]; };

// Load one 16x32 (f16) WMMA A/B fragment from row-major f16 memory.
// Lane L<16  : row L,    halfs k0+{0..7}   and k0+{16..23}
// Lane L>=16 : row L-16, halfs k0+{8..15}  and k0+{24..31}
// Both 16-byte chunks are contiguous -> two global_load_b128 per lane.
__device__ __forceinline__ v16h load_frag16(const _Float16* __restrict__ base,
                                            int row_stride, int k0) {
  const int lane = threadIdx.x & 31;
  const int r    = lane & 15;
  const int koff = k0 + ((lane >> 4) << 3);
  const _Float16* p = base + (size_t)r * row_stride + koff;
  FragU u;
  u.f4[0] = *(const f32x4*)(p);        // halfs koff .. koff+7
  u.f4[1] = *(const f32x4*)(p + 16);   // halfs koff+16 .. koff+23
  return u.h;
}

// ---------------------------------------------------------------------------
// Kernel 1: f32 -> f16 copies: xh (row-major) and xt (per-batch transposed)
// ---------------------------------------------------------------------------
__global__ void __launch_bounds__(256)
convert_kernel(const float* __restrict__ x,
               _Float16* __restrict__ xh, _Float16* __restrict__ xt) {
  const int idx = blockIdx.x * 256 + threadIdx.x;     // over B*S*F = 524288
  const float v = x[idx];
  const _Float16 h = (_Float16)v;
  xh[idx] = h;
  const int f = idx & (F_ - 1);
  const int s = (idx >> 7) & (S_ - 1);
  const int b = idx >> 16;
  xt[((size_t)b * F_ + f) * S_ + s] = h;
}

// ---------------------------------------------------------------------------
// Kernel 2: fused attention -> ctx[b,i,f] (f32, 2 MB).  One wave (32 threads)
// per (batch, 16-query tile).  256 waves total, 256 WMMAs each.
// ---------------------------------------------------------------------------
__global__ void __launch_bounds__(32)
attn_ctx_kernel(const _Float16* __restrict__ xh,
                const _Float16* __restrict__ xt,
                float* __restrict__ ctx) {
  const int b     = blockIdx.x >> 5;          // 32 i-tiles per batch
  const int itile = blockIdx.x & 31;
  const int i0    = itile * 16;
  const int lane  = threadIdx.x & 31;

  const _Float16* xb  = xh + (size_t)b * S_ * F_;   // X   (S x F), row-major
  const _Float16* xtb = xt + (size_t)b * F_ * S_;   // X^T (F x S), row-major

  // Query-tile B fragments (K = feature dim, 4 chunks of 32): loaded once.
  v16h bq[4];
#pragma unroll
  for (int kk = 0; kk < 4; ++kk)
    bq[kk] = load_frag16(xb + (size_t)i0 * F_, F_, kk * 32);

  v8f ctxacc[8];
#pragma unroll
  for (int t = 0; t < 8; ++t) ctxacc[t] = (v8f){};

  float dsum = 0.f;   // per-lane partial of column sums of exp(P)

  for (int s0 = 0; s0 < S_; s0 += 32) {
    // ---- scores P[s, i] = X[s,:] . X[i,:], two 16-row s-subtiles ----
    v8f p0 = (v8f){}, p1 = (v8f){};
#pragma unroll
    for (int kk = 0; kk < 4; ++kk) {
      v16h a0 = load_frag16(xb + (size_t)s0 * F_, F_, kk * 32);
      p0 = __builtin_amdgcn_wmma_f32_16x16x32_f16(false, a0, false, bq[kk],
                                                  (short)0, p0, false, false);
    }
#pragma unroll
    for (int kk = 0; kk < 4; ++kk) {
      v16h a1 = load_frag16(xb + (size_t)(s0 + 16) * F_, F_, kk * 32);
      p1 = __builtin_amdgcn_wmma_f32_16x16x32_f16(false, a1, false, bq[kk],
                                                  (short)0, p1, false, false);
    }

    // ---- exp, denominator partials, and repack as A-fragment of exp(P)^T.
    // D-layout (lane = i, VGPR = s) coincides exactly with the A-fragment
    // layout of the transpose (lane = i row, VGPR pairs = K = s): no shuffles.
    v16h ap;
#pragma unroll
    for (int r = 0; r < 8; ++r) {
      const float e0 = __expf(p0[r]);
      const float e1 = __expf(p1[r]);
      dsum += e0 + e1;
      ap[r]     = (_Float16)e0;   // K = s0 +  r(+8)   half of chunk
      ap[8 + r] = (_Float16)e1;   // K = s0 + 16 + r(+8)
    }

    // ---- ctx += exp(P)^T (16 x 32) * X[s0..s0+31, :]  (8 f-tiles) ----
#pragma unroll
    for (int ft = 0; ft < 8; ++ft) {
      v16h bt = load_frag16(xtb + (size_t)(ft * 16) * S_, S_, s0);
      ctxacc[ft] = __builtin_amdgcn_wmma_f32_16x16x32_f16(
          false, ap, false, bt, (short)0, ctxacc[ft], false, false);
    }
  }

  // Total denominator per query i: combine half-wave partials, then fetch the
  // reciprocal for each of this lane's 8 accumulator row indices.
  dsum += __shfl_xor(dsum, 16, 32);
  float rec[8];
#pragma unroll
  for (int r = 0; r < 8; ++r) {
    const int src = r + ((lane >> 4) << 3);          // row m this VGPR maps to
    rec[r] = 1.0f / (__shfl(dsum, src, 32) + 1e-7f);
  }

  // Scatter scaled ctx (D-layout: lane = f-in-tile, VGPR = i-row).
  float* cb = ctx + ((size_t)b * S_ + i0) * F_;
#pragma unroll
  for (int ft = 0; ft < 8; ++ft) {
#pragma unroll
    for (int r = 0; r < 8; ++r) {
      const int m = r + ((lane >> 4) << 3);
      cb[(size_t)m * F_ + ft * 16 + (lane & 15)] = ctxacc[ft][r] * rec[r];
    }
  }
}

// ---------------------------------------------------------------------------
// Kernel 3: bandwidth-bound expansion: out[row] = [ x[row] | ctx[row] x 128 ].
// ctx reads are L2-resident (2 MB); 270 MB of output goes out as NT b128
// stores.  One block per output row; stride 256 is a multiple of 32 so each
// thread's ctx float4 index is fixed -> load once, store ~16 times.
// ---------------------------------------------------------------------------
__global__ void __launch_bounds__(256)
expand_kernel(const float* __restrict__ x, const float* __restrict__ ctx,
              float* __restrict__ out) {
  const int row = blockIdx.x;                  // b*S + i, 4096 rows
  const int t   = threadIdx.x;
  const f32x4* x4 = (const f32x4*)x   + (size_t)row * (F_ / 4);
  const f32x4* c4 = (const f32x4*)ctx + (size_t)row * (F_ / 4);
  f32x4*       o4 = (f32x4*)out       + (size_t)row * (OUTW / 4);

  const f32x4 cv = c4[t & 31];
  const int n4 = OUTW / 4;                     // 4128
  for (int j = t; j < n4; j += 256) {
    f32x4 v = cv;
    if (j < F_ / 4) v = x4[j];                 // first 128 floats = x row
    __builtin_nontemporal_store(v, &o4[j]);
  }
}

// ---------------------------------------------------------------------------
extern "C" void kernel_launch(void* const* d_in, const int* in_sizes, int n_in,
                              void* d_out, int out_size, void* d_ws, size_t ws_size,
                              hipStream_t stream) {
  const float* x = (const float*)d_in[0];
  float* out = (float*)d_out;

  char* ws = (char*)d_ws;
  const size_t nelem = (size_t)B_ * S_ * F_;           // 524288
  _Float16* xh  = (_Float16*)(ws);                     // 1 MB
  _Float16* xt  = (_Float16*)(ws + nelem * 2);         // 1 MB
  float*    ctx = (float*)   (ws + nelem * 4);         // 2 MB

  convert_kernel<<<(int)(nelem / 256), 256, 0, stream>>>(x, xh, xt);
  attn_ctx_kernel<<<B_ * (S_ / 16), 32, 0, stream>>>(xh, xt, ctx);
  expand_kernel<<<B_ * S_, 256, 0, stream>>>(x, ctx, out);
}
